// RWKVBlock_79053168050654
// MI455X (gfx1250) — matrix operand order repaired and checked by reading, not compiled, not run on hardware
//
#include <hip/hip_runtime.h>
#include <hip/hip_bf16.h>
#include <stdint.h>

typedef __attribute__((ext_vector_type(16))) __bf16 v16bf;
typedef __attribute__((ext_vector_type(8)))  __bf16 v8bf;
typedef __attribute__((ext_vector_type(8)))  float  v8f;

#define BDIM  4
#define TDIM  4096
#define CDIM  1024
#define MROWS (BDIM * TDIM)
#define LN_EPS 1e-5f

// ---------------------------------------------------------------------------
// CDNA5 async global->LDS copy (ASYNCcnt-tracked, no VGPR data path)
// ---------------------------------------------------------------------------
__device__ __forceinline__ void async_b128(uint32_t lds_off, const void* gaddr) {
    asm volatile("global_load_async_to_lds_b128 %0, %1, off"
                 :: "v"(lds_off), "v"(gaddr) : "memory");
}
template <int N>
__device__ __forceinline__ void wait_asynccnt() {
    asm volatile("s_wait_asynccnt %0" :: "n"(N) : "memory");
}

// ---------------------------------------------------------------------------
// Weight prep: W (KxN, f32, row-major) -> Wt (NxK, bf16, row-major).
// 64x64 tiles through LDS; coalesced on both sides. Weights stay L2-resident.
// ---------------------------------------------------------------------------
__global__ __launch_bounds__(256) void transpose_convert_kernel(
    const float* __restrict__ W, __bf16* __restrict__ Wt, int K, int N)
{
    __shared__ __bf16 tile[64][72];
    const int bn = blockIdx.x * 64;       // N dim
    const int bk = blockIdx.y * 64;       // K dim
    const int r = threadIdx.x >> 6;       // 0..3
    const int c = threadIdx.x & 63;       // 0..63
#pragma unroll
    for (int i = 0; i < 16; ++i) {
        const int k = r + 4 * i;
        tile[c][k] = (__bf16)W[(size_t)(bk + k) * N + (bn + c)];
    }
    __syncthreads();
#pragma unroll
    for (int i = 0; i < 16; ++i) {
        const int n = r + 4 * i;
        Wt[(size_t)(bn + n) * K + (bk + c)] = tile[n][c];
    }
}

// ---------------------------------------------------------------------------
// LayerNorm over C + time-shift -> bf16:
//   hs[b, t+1, :] = LN(x[b, t, :]) * g + b  (t < T-1);  hs[b, 0, :] = time_shift
// ---------------------------------------------------------------------------
__global__ __launch_bounds__(256) void ln_shift_kernel(
    const float* __restrict__ x, const float* __restrict__ g,
    const float* __restrict__ bta, const float* __restrict__ ts,
    __bf16* __restrict__ hs)
{
    const int row  = blockIdx.x;        // b*T + t
    const int t    = row % TDIM;
    const int tid  = threadIdx.x;
    const int lane = tid & 31;
    const int wid  = tid >> 5;
    const float* xr = x + (size_t)row * CDIM;

    float xe[4];
    float s = 0.f;
#pragma unroll
    for (int i = 0; i < 4; ++i) { xe[i] = xr[tid + 256 * i]; s += xe[i]; }
#pragma unroll
    for (int o = 16; o > 0; o >>= 1) s += __shfl_xor(s, o, 32);

    __shared__ float red[8];
    if (lane == 0) red[wid] = s;
    __syncthreads();
    float acc8 = (lane < 8) ? red[lane] : 0.f;
#pragma unroll
    for (int o = 4; o > 0; o >>= 1) acc8 += __shfl_xor(acc8, o, 32);
    const float mean = __shfl(acc8, 0, 32) * (1.0f / CDIM);

    float s2 = 0.f;
#pragma unroll
    for (int i = 0; i < 4; ++i) { float d = xe[i] - mean; s2 += d * d; }
#pragma unroll
    for (int o = 16; o > 0; o >>= 1) s2 += __shfl_xor(s2, o, 32);
    __syncthreads();
    if (lane == 0) red[wid] = s2;
    __syncthreads();
    float acc8b = (lane < 8) ? red[lane] : 0.f;
#pragma unroll
    for (int o = 4; o > 0; o >>= 1) acc8b += __shfl_xor(acc8b, o, 32);
    const float var  = __shfl(acc8b, 0, 32) * (1.0f / CDIM);
    const float rstd = rsqrtf(var + LN_EPS);

    __bf16* dst = (t + 1 < TDIM) ? (hs + (size_t)(row + 1) * CDIM) : nullptr;
#pragma unroll
    for (int i = 0; i < 4; ++i) {
        const int c = tid + 256 * i;
        if (dst) dst[c] = (__bf16)((xe[i] - mean) * rstd * g[c] + bta[c]);
    }
    if (t == 0) {
        __bf16* d0 = hs + (size_t)row * CDIM;
#pragma unroll
        for (int i = 0; i < 4; ++i) { const int c = tid + 256 * i; d0[c] = (__bf16)ts[c]; }
    }
}

// ---------------------------------------------------------------------------
// NT bf16 WMMA GEMM: C(MxN, f32 acc) = A(MxK, bf16 row-major) * Bt(NxK, bf16)^T
// Block tile 128x128x64, 8 waves (wave = 32x64 patch = 2x4 WMMA tiles,
// 16 v_wmma per iter). Both tiles staged by async global->LDS b128 copies,
// double buffered on ASYNCcnt.
// ---------------------------------------------------------------------------
enum { EPI_F32 = 0, EPI_RELUSQ_BF16 = 1, EPI_FINAL = 2 };

template <int EPI>
__global__ __launch_bounds__(256) void gemm_wmma_kernel(
    const __bf16* __restrict__ A, const __bf16* __restrict__ Bt,
    int M, int N, int K,
    float* __restrict__ outF, __bf16* __restrict__ outBF,
    const float* __restrict__ rr, const float* __restrict__ xb)
{
    constexpr int BM = 128, BN = 128, BK = 64;
    constexpr int LDT = BK + 8;                 // padded LDS stride (elements)
    __shared__ __align__(16) __bf16 As[2][BM * LDT];
    __shared__ __align__(16) __bf16 Bs[2][BN * LDT];

    const int tid  = threadIdx.x;
    const int lane = tid & 31;
    const int wid  = tid >> 5;
    const int wm   = wid & 3;                   // 4 waves x 32 rows
    const int wn   = wid >> 2;                  // 2 waves x 64 cols
    const int bm   = blockIdx.y * BM;
    const int bn   = blockIdx.x * BN;

    // staging geometry: thread owns 64 contiguous bytes of one 128B tile row
    const int srow = tid >> 1;                  // 0..127
    const int scol = (tid & 1) * 32;            // element offset 0 / 32

    // fragment geometry (ISA bf16 wave32 layouts)
    const int kbA = (lane >= 16) ? 8 : 0;       // A: K = {kb..kb+7, kb+16..kb+23}
    const int rA  = lane & 15;
    const int kbB = (lane >= 16) ? 8 : 0;       // B: K = {kb..kb+7, kb+16..kb+23}
    const int nB  = lane & 15;

    v8f acc[2][4] = {};

    auto stageA = [&](int buf, int k0) {
        const __bf16* g = A + (size_t)(bm + srow) * K + (k0 + scol);
        uint32_t l = (uint32_t)(uintptr_t)&As[buf][srow * LDT + scol];
#pragma unroll
        for (int j = 0; j < 4; ++j) async_b128(l + 16u * j, g + 8 * j);
    };
    auto stageB = [&](int buf, int k0) {
        const __bf16* g = Bt + (size_t)(bn + srow) * K + (k0 + scol);
        uint32_t l = (uint32_t)(uintptr_t)&Bs[buf][srow * LDT + scol];
#pragma unroll
        for (int j = 0; j < 4; ++j) async_b128(l + 16u * j, g + 8 * j);
    };

    const int niter = K / BK;
    stageA(0, 0);
    stageB(0, 0);

    for (int it = 0; it < niter; ++it) {
        const int cur = it & 1;
        if (it + 1 < niter) {
            stageA(cur ^ 1, (it + 1) * BK);
            stageB(cur ^ 1, (it + 1) * BK);
            wait_asynccnt<8>();                 // older batch (cur) complete
        } else {
            wait_asynccnt<0>();
        }
        __syncthreads();

#pragma unroll
        for (int ks = 0; ks < BK; ks += 32) {
            v16bf afrag[2], bfrag[4];
#pragma unroll
            for (int mi = 0; mi < 2; ++mi) {
                const __bf16* ap = &As[cur][(wm * 32 + mi * 16 + rA) * LDT + ks + kbA];
                v8bf lo = *(const v8bf*)ap;
                v8bf hi = *(const v8bf*)(ap + 16);
#pragma unroll
                for (int j = 0; j < 8; ++j) { afrag[mi][j] = lo[j]; afrag[mi][8 + j] = hi[j]; }
            }
#pragma unroll
            for (int ni = 0; ni < 4; ++ni) {
                const __bf16* bp = &Bs[cur][(wn * 64 + ni * 16 + nB) * LDT + ks + kbB];
                v8bf lo = *(const v8bf*)bp;
                v8bf hi = *(const v8bf*)(bp + 16);
#pragma unroll
                for (int j = 0; j < 8; ++j) { bfrag[ni][j] = lo[j]; bfrag[ni][8 + j] = hi[j]; }
            }
#pragma unroll
            for (int mi = 0; mi < 2; ++mi)
#pragma unroll
                for (int ni = 0; ni < 4; ++ni)
                    acc[mi][ni] = __builtin_amdgcn_wmma_f32_16x16x32_bf16(
                        false, afrag[mi], false, bfrag[ni],
                        (short)0, acc[mi][ni], false, false);
        }
        __syncthreads();
    }

    // epilogue (C/D layout: lane<16 -> N=lane, M=r; lane>=16 -> N=lane-16, M=r+8)
    const int mBase = (lane >= 16) ? 8 : 0;
    const int col0  = lane & 15;
#pragma unroll
    for (int mi = 0; mi < 2; ++mi)
#pragma unroll
        for (int ni = 0; ni < 4; ++ni)
#pragma unroll
            for (int rI = 0; rI < 8; ++rI) {
                const int row = bm + wm * 32 + mi * 16 + mBase + rI;
                const int col = bn + wn * 64 + ni * 16 + col0;
                const size_t idx = (size_t)row * N + col;
                const float a = acc[mi][ni][rI];
                if constexpr (EPI == EPI_F32) {
                    outF[idx] = a;
                } else if constexpr (EPI == EPI_RELUSQ_BF16) {
                    float tpos = fmaxf(a, 0.f);
                    outBF[idx] = (__bf16)(tpos * tpos);
                } else {
                    float sr = 1.f / (1.f + __expf(-rr[idx]));
                    outF[idx] = xb[idx] + sr * a;
                }
            }
}

// ---------------------------------------------------------------------------
// WKV recurrence fused with x += sigmoid(r)*y. One lane per (b, c) channel.
// ---------------------------------------------------------------------------
__global__ __launch_bounds__(256) void wkv_kernel(
    const float* __restrict__ kbuf, const float* __restrict__ vbuf,
    const float* __restrict__ rbuf, const float* __restrict__ x,
    const float* __restrict__ wdec, const float* __restrict__ ufirst,
    float* __restrict__ xmid)
{
    const int idx = blockIdx.x * blockDim.x + threadIdx.x;   // b*C + c
    if (idx >= BDIM * CDIM) return;
    const int b = idx / CDIM, c = idx % CDIM;
    const float w = wdec[c], u = ufirst[c];

    float aa = 0.f, bb = 1.f, pp = -1e38f;
    size_t off = (size_t)b * TDIM * CDIM + c;
    for (int t = 0; t < TDIM; ++t, off += CDIM) {
        const float kk = kbuf[off], vv = vbuf[off];
        const float uk = u + kk;
        float p  = fmaxf(pp, uk);
        float e1 = __expf(pp - p);
        float e2 = __expf(uk - p);
        float y  = (e1 * aa + e2 * vv) / (e1 * bb + e2);
        const float wk = w + kk;
        float pu  = fmaxf(pp + w, wk);
        float e1u = __expf(pp + w - pu);
        float e2u = __expf(wk - pu);
        aa = e1u * aa + e2u * vv;
        bb = e1u * bb + e2u;
        pp = pu + __logf(bb);
        const float sr = 1.f / (1.f + __expf(-rbuf[off]));
        xmid[off] = x[off] + sr * y;
    }
}

// ---------------------------------------------------------------------------
extern "C" void kernel_launch(void* const* d_in, const int* in_sizes, int n_in,
                              void* d_out, int out_size, void* d_ws, size_t ws_size,
                              hipStream_t stream) {
    const float* x   = (const float*)d_in[0];
    const float* td  = (const float*)d_in[1];   // time_decay (w)
    const float* tf  = (const float*)d_in[2];   // time_first (u)
    const float* ts  = (const float*)d_in[3];
    const float* Wk  = (const float*)d_in[4];
    const float* Wv  = (const float*)d_in[5];
    const float* Wr  = (const float*)d_in[6];
    const float* Wck = (const float*)d_in[7];
    const float* Wcv = (const float*)d_in[8];
    const float* Wcr = (const float*)d_in[9];
    const float* g1  = (const float*)d_in[10];
    const float* b1  = (const float*)d_in[11];
    const float* g2  = (const float*)d_in[12];
    const float* b2  = (const float*)d_in[13];
    float* out = (float*)d_out;

    char* ws = (char*)d_ws;
    const size_t MC  = (size_t)MROWS * CDIM;        // 16M elements
    const size_t CC2 = (size_t)CDIM * CDIM * 2;     // 2MB (bf16 CxC)

    __bf16* hs    = (__bf16*)(ws);                          // 2*MC
    float*  kbuf  = (float*)(ws + MC * 2);                  // 4*MC
    float*  vbuf  = (float*)(ws + MC * 2 + MC * 4);         // 4*MC
    float*  rbuf  = (float*)(ws + MC * 2 + MC * 8);         // 4*MC
    float*  xmid  = (float*)(ws + MC * 2 + MC * 12);        // 4*MC
    __bf16* kkbuf = (__bf16*)(ws + MC * 2);                 // aliases k+v (dead)
    float*  rrbuf = rbuf;                                   // aliases r (dead)

    char* wp = ws + MC * 18;                                // transposed bf16 weights
    __bf16* wtk  = (__bf16*)(wp);
    __bf16* wtv  = (__bf16*)(wp + CC2);
    __bf16* wtr  = (__bf16*)(wp + 2 * CC2);
    __bf16* wtcr = (__bf16*)(wp + 3 * CC2);
    __bf16* wtck = (__bf16*)(wp + 4 * CC2);                 // [4C][C]  (4*CC2)
    __bf16* wtcv = (__bf16*)(wp + 8 * CC2);                 // [C][4C]  (4*CC2)

    const dim3 blk(256);

    // weight prep: f32 [K][N] -> bf16 [N][K]
    transpose_convert_kernel<<<dim3(CDIM/64, CDIM/64), blk, 0, stream>>>(Wk,  wtk,  CDIM, CDIM);
    transpose_convert_kernel<<<dim3(CDIM/64, CDIM/64), blk, 0, stream>>>(Wv,  wtv,  CDIM, CDIM);
    transpose_convert_kernel<<<dim3(CDIM/64, CDIM/64), blk, 0, stream>>>(Wr,  wtr,  CDIM, CDIM);
    transpose_convert_kernel<<<dim3(CDIM/64, CDIM/64), blk, 0, stream>>>(Wcr, wtcr, CDIM, CDIM);
    transpose_convert_kernel<<<dim3(4*CDIM/64, CDIM/64), blk, 0, stream>>>(Wck, wtck, CDIM, 4*CDIM);
    transpose_convert_kernel<<<dim3(CDIM/64, 4*CDIM/64), blk, 0, stream>>>(Wcv, wtcv, 4*CDIM, CDIM);

    const dim3 gC(CDIM / 128, MROWS / 128);        // N = C
    const dim3 g4C(4 * CDIM / 128, MROWS / 128);   // N = 4C

    // time mixing
    ln_shift_kernel<<<MROWS, blk, 0, stream>>>(x, g1, b1, ts, hs);
    gemm_wmma_kernel<EPI_F32><<<gC, blk, 0, stream>>>(
        hs, wtk, MROWS, CDIM, CDIM, kbuf, nullptr, nullptr, nullptr);
    gemm_wmma_kernel<EPI_F32><<<gC, blk, 0, stream>>>(
        hs, wtv, MROWS, CDIM, CDIM, vbuf, nullptr, nullptr, nullptr);
    gemm_wmma_kernel<EPI_F32><<<gC, blk, 0, stream>>>(
        hs, wtr, MROWS, CDIM, CDIM, rbuf, nullptr, nullptr, nullptr);
    wkv_kernel<<<(BDIM * CDIM) / 256, blk, 0, stream>>>(
        kbuf, vbuf, rbuf, x, td, tf, xmid);

    // channel mixing
    ln_shift_kernel<<<MROWS, blk, 0, stream>>>(xmid, g2, b2, ts, hs);
    gemm_wmma_kernel<EPI_RELUSQ_BF16><<<g4C, blk, 0, stream>>>(
        hs, wtck, MROWS, 4 * CDIM, CDIM, nullptr, kkbuf, nullptr, nullptr);
    gemm_wmma_kernel<EPI_F32><<<gC, blk, 0, stream>>>(
        hs, wtcr, MROWS, CDIM, CDIM, rrbuf, nullptr, nullptr, nullptr);
    gemm_wmma_kernel<EPI_FINAL><<<gC, blk, 0, stream>>>(
        kkbuf, wtcv, MROWS, CDIM, 4 * CDIM, out, nullptr, rrbuf, xmid);
}
